// bi_mLSTMCell_79723182948408
// MI455X (gfx1250) — matrix operand order repaired and checked by reading, not compiled, not run on hardware
//
#include <hip/hip_runtime.h>
#include <math.h>

// ---------------------------------------------------------------------------
// bi-mLSTM cell for MI455X (gfx1250).
// bf16 WMMA (v_wmma_f32_16x16x32_bf16) + async global->LDS double-buffered
// pipeline (GLOBAL_LOAD_ASYNC_TO_LDS_B128 / s_wait_asynccnt).
// ---------------------------------------------------------------------------

typedef unsigned short ushort_t;
typedef __attribute__((ext_vector_type(16))) __bf16 v16bf;
typedef __attribute__((ext_vector_type(8)))  float  v8f;

union Frag {
    v16bf v;
    uint4 u[2];
};

#define B_    2
#define D_    1024
#define H_    8
#define SL_   512
#define HS_   512
#define DK_   64
#define M_    2048    /* B_*1024 rows on each side      */
#define NC_   1024    /* cell-local N (== D, symmetric) */
#define SCALE_ 0.125f /* DK^-0.5 */

// ---------------- async global->LDS (CDNA5) ----------------
#if __has_builtin(__builtin_amdgcn_global_load_async_to_lds_b128) && \
    __has_builtin(__builtin_amdgcn_s_wait_asynccnt)
#define USE_ASYNC 1
#else
#define USE_ASYNC 0
#endif

#if USE_ASYNC
typedef __attribute__((ext_vector_type(4))) int i4v;
typedef __attribute__((address_space(1))) i4v* as1_i4v;
typedef __attribute__((address_space(3))) i4v* as3_i4v;
#endif

__device__ __forceinline__ void async_copy_b128(void* lds, const void* g) {
#if USE_ASYNC
    __builtin_amdgcn_global_load_async_to_lds_b128(
        (as1_i4v)(unsigned long long)g, (as3_i4v)lds, 0, 0);
#else
    *(uint4*)lds = *(const uint4*)g;
#endif
}
__device__ __forceinline__ void async_wait_stage() {   // wait: prev stage done
#if USE_ASYNC
    __builtin_amdgcn_s_wait_asynccnt(4);
#endif
}
__device__ __forceinline__ void async_wait_all() {
#if USE_ASYNC
    __builtin_amdgcn_s_wait_asynccnt(0);
#endif
}

// ---------------- helpers ----------------

__device__ __forceinline__ ushort_t f2bf(float f) {
    unsigned u = __builtin_bit_cast(unsigned, f);
    unsigned r = (u + 0x7FFFu + ((u >> 16) & 1u)) >> 16;
    return (ushort_t)r;
}

__device__ __forceinline__ float bf2f(ushort_t h) {
    unsigned u = ((unsigned)h) << 16;
    return __builtin_bit_cast(float, u);
}

__device__ __forceinline__ v8f wmma_bf16(const Frag& a, const Frag& b, v8f c) {
    return __builtin_amdgcn_wmma_f32_16x16x32_bf16(
        false, a.v, false, b.v, (short)0, c, false, false);
}

// ---------------------------------------------------------------------------
// Weight-norm prep: w_bf16[row][:] = g[row] * v[row][:] / ||v[row]||_2
// (g == nullptr -> plain fp32->bf16 convert).  in_f == 512 everywhere.
// ---------------------------------------------------------------------------
__global__ __launch_bounds__(256)
void wn_prep_kernel(const float* __restrict__ v, const float* __restrict__ g,
                    ushort_t* __restrict__ w, int in_f) {
    __shared__ float red[256];
    const int row = blockIdx.x;
    const float* vr = v + (long)row * in_f;
    float s = 0.f;
    for (int c = threadIdx.x; c < in_f; c += 256) { float x = vr[c]; s += x * x; }
    red[threadIdx.x] = s;
    __syncthreads();
    for (int off = 128; off > 0; off >>= 1) {
        if (threadIdx.x < off) red[threadIdx.x] += red[threadIdx.x + off];
        __syncthreads();
    }
    const float scale = (g != nullptr) ? (g[row] * rsqrtf(red[0])) : 1.0f;
    ushort_t* wr = w + (long)row * in_f;
    for (int c = threadIdx.x; c < in_f; c += 256) wr[c] = f2bf(vr[c] * scale);
}

// fp32 -> bf16 elementwise
__global__ __launch_bounds__(256)
void f2bf_kernel(const float* __restrict__ x, ushort_t* __restrict__ y, int n) {
    int i = blockIdx.x * 256 + threadIdx.x;
    if (i < n) y[i] = f2bf(x[i]);
}

// split gates (M_ x 1536) -> gi_bf16, gf_bf16 (M_ x 512), osig fp32
__global__ __launch_bounds__(256)
void split_gates_kernel(const float* __restrict__ gates,
                        ushort_t* __restrict__ gi, ushort_t* __restrict__ gf,
                        float* __restrict__ osig, int n) {
    int i = blockIdx.x * 256 + threadIdx.x;
    if (i >= n) return;
    int row = i >> 9;
    int c   = i & 511;
    const float* gr = gates + (long)row * 1536;
    gi[i] = f2bf(gr[c]);
    gf[i] = f2bf(gr[512 + c]);
    float x = gr[1024 + c];
    osig[i] = 1.f / (1.f + expf(-x));
}

// ---------------------------------------------------------------------------
// bf16 WMMA GEMM:  out = epilogue( A(MxK) @ W(NxK)^T + bias )
//   block = 256 threads (8 waves), block tile 128x128, wave tile 32x64.
//   Double-buffered LDS tiles filled with async global->LDS copies.
// Epilogue modes:
//   0: fp32 row-major; optional accumulate, optional elementwise multiplier.
//   2: tanh    -> bf16, permuted (b,g,r,j) store        [i gate]
//   3: sigmoid -> bf16, permuted store                  [f gate]
//   4: (x+bias)*scale -> bf16, permuted store           [q, k]
//   5: (x+bias)*scale -> bf16, transposed-permuted (b,g,j,r) store [v^T]
// Requires M%128==0, N%128==0, K%32==0 (true for all calls here).
// ---------------------------------------------------------------------------
__global__ __launch_bounds__(256)
void gemm_wmma_kernel(const ushort_t* __restrict__ A, int lda,
                      const ushort_t* __restrict__ W, int ldw,
                      int Mdim, int Ndim, int Kdim,
                      const float* __restrict__ bias,
                      const float* __restrict__ mul,
                      float out_scale, int mode, int accumulate,
                      float* __restrict__ outF, ushort_t* __restrict__ outB,
                      int ldo, int G, int Rrows) {
    (void)Mdim; (void)Ndim;
    __shared__ ushort_t At[2][128][40];   // 128x32 tiles, padded rows (16B-aligned)
    __shared__ ushort_t Wt[2][128][40];

    const int tid  = threadIdx.x;
    const int lane = tid & 31;
    const int wv   = tid >> 5;       // 0..7
    const int wr   = wv & 3;         // wave row-group (32 rows)
    const int wc   = wv >> 2;        // wave col-group (64 cols)
    const int m16  = lane & 15;
    const int kh   = lane >> 4;      // 0/1
    const int rowA0 = blockIdx.y * 128;
    const int colW0 = blockIdx.x * 128;

    // issue one K-stage of async copies (4 ops per thread -> ASYNCcnt += 4/wave)
    auto stage = [&](int buf, int k0) {
#pragma unroll
        for (int it = 0; it < 2; ++it) {
            int vid = tid + it * 256;        // 0..511
            int r   = vid >> 2;              // 0..127
            int cv  = (vid & 3) * 8;
            async_copy_b128(&At[buf][r][cv], &A[(long)(rowA0 + r) * lda + k0 + cv]);
            async_copy_b128(&Wt[buf][r][cv], &W[(long)(colW0 + r) * ldw + k0 + cv]);
        }
    };

    v8f acc[2][4];
    const v8f vz = {0.f, 0.f, 0.f, 0.f, 0.f, 0.f, 0.f, 0.f};
#pragma unroll
    for (int i = 0; i < 2; ++i)
#pragma unroll
        for (int j = 0; j < 4; ++j) acc[i][j] = vz;

    stage(0, 0);
    int buf = 0;
    for (int k0 = 0; k0 < Kdim; k0 += 32, buf ^= 1) {
        const bool hasNext = (k0 + 32 < Kdim);
        if (hasNext) stage(buf ^ 1, k0 + 32);          // overlap with compute
        if (k0 + 64 < Kdim) {                          // L2 prefetch 2 stages out
            __builtin_prefetch(&A[(long)(rowA0 + (tid & 127)) * lda + k0 + 64], 0, 1);
            __builtin_prefetch(&W[(long)(colW0 + (tid & 127)) * ldw + k0 + 64], 0, 1);
        }
        if (hasNext) async_wait_stage(); else async_wait_all();
        __syncthreads();                               // stage `buf` visible to all

        Frag af[2], wf[4];
#pragma unroll
        for (int i = 0; i < 2; ++i) {
            // A operand: lane(m,kh): K in {kh*8..+7} u {16+kh*8..+7}
            int ar = wr * 32 + i * 16 + m16;
            af[i].u[0] = *(const uint4*)&At[buf][ar][kh * 8];
            af[i].u[1] = *(const uint4*)&At[buf][ar][16 + kh * 8];
        }
#pragma unroll
        for (int j = 0; j < 4; ++j) {
            // B operand: lane(n,kh): K in {kh*8..+7} u {16+kh*8..+7}
            int wrow = wc * 64 + j * 16 + m16;
            wf[j].u[0] = *(const uint4*)&Wt[buf][wrow][kh * 8];
            wf[j].u[1] = *(const uint4*)&Wt[buf][wrow][16 + kh * 8];
        }
#pragma unroll
        for (int i = 0; i < 2; ++i)
#pragma unroll
            for (int j = 0; j < 4; ++j)
                acc[i][j] = wmma_bf16(af[i], wf[j], acc[i][j]);
        __syncthreads();                               // done reading `buf`
    }

    // epilogue: C/D layout: vgpr r, lane -> (m = r + 8*kh, n = m16)
#pragma unroll
    for (int i = 0; i < 2; ++i)
#pragma unroll
        for (int j = 0; j < 4; ++j)
#pragma unroll
            for (int r = 0; r < 8; ++r) {
                int mrow = rowA0 + wr * 32 + i * 16 + r + 8 * kh;
                int ncol = colW0 + wc * 64 + j * 16 + m16;
                float x = acc[i][j][r];
                if (bias) x += bias[ncol];
                if (mode == 0) {
                    long o = (long)mrow * ldo + ncol;
                    if (accumulate) x += outF[o];
                    if (mul) x *= mul[o];
                    outF[o] = x;
                } else {
                    if (mode == 2)      x = tanhf(x);
                    else if (mode == 3) x = 1.f / (1.f + expf(-x));
                    else                x *= out_scale;          // modes 4,5
                    int gg = ncol / G, jj = ncol - gg * G;
                    int bb = mrow / Rrows, rr = mrow - bb * Rrows;
                    long dst = (mode == 5)
                        ? ((((long)bb * H_ + gg) * G + jj) * Rrows + rr)   // (B,H,G,R)
                        : ((((long)bb * H_ + gg) * Rrows + rr) * G + jj);  // (B,H,R,G)
                    outB[dst] = f2bf(x);
                }
            }
}

// ---------------------------------------------------------------------------
// Fused attention / state update / value contraction, per (b,h):
//   attn = q' @ k^T            (SCALE folded into q')
//   Cn   = C * f + attn * i    -> streamed to d_out (fp32, (B,H,R,NC))
//   out  = Cn @ v              -> bf16 in (B,R,H*DK) layout for o-projection
// v is pre-transposed to (B,H,DK,R) so its B-fragments come straight from
// global memory; the only LDS use is the wave-private Cn re-layout tile,
// so the inner loop has no barriers.
// ---------------------------------------------------------------------------
__global__ __launch_bounds__(128)
void attn_fused_kernel(const ushort_t* __restrict__ q,   // (B,H,R,DK) bf16, pre-scaled
                       const ushort_t* __restrict__ k,   // (B,H,R,DK) bf16
                       const ushort_t* __restrict__ vT,  // (B,H,DK,R) bf16
                       const ushort_t* __restrict__ iact,// (B,H,R,NC) bf16
                       const ushort_t* __restrict__ fact,// (B,H,R,NC) bf16
                       const float*    __restrict__ C,   // (B,H,R,NC) fp32
                       float*          __restrict__ Cn,  // (B,H,R,NC) fp32 (d_out)
                       ushort_t*       __restrict__ outB)// (B,R,H*DK)  bf16
{
    __shared__ ushort_t cnS[4][16][40];   // per-wave Cn tile 16 x 32 (no barriers)

    const int tid  = threadIdx.x;
    const int lane = tid & 31;
    const int wv   = tid >> 5;
    const int m16  = lane & 15;
    const int kh   = lane >> 4;
    const int bh   = blockIdx.y;          // b*H + h
    const int b    = bh >> 3;
    const int hh   = bh & 7;
    const int d0   = blockIdx.x * 64;
    const long qkBase = (long)bh * NC_ * DK_;
    const long ifBase = (long)bh * NC_ * NC_;

    // preload q A-fragments for this wave's 16 rows (K = 0..63 in two chunks)
    Frag qf[2];
    {
        int dr = d0 + wv * 16 + m16;
        const ushort_t* qp = q + qkBase + (long)dr * DK_;
        qf[0].u[0] = *(const uint4*)&qp[kh * 8];
        qf[0].u[1] = *(const uint4*)&qp[16 + kh * 8];
        qf[1].u[0] = *(const uint4*)&qp[32 + kh * 8];
        qf[1].u[1] = *(const uint4*)&qp[48 + kh * 8];
    }

    const v8f vz = {0.f, 0.f, 0.f, 0.f, 0.f, 0.f, 0.f, 0.f};
    v8f outAcc[4];
#pragma unroll
    for (int cb = 0; cb < 4; ++cb) outAcc[cb] = vz;

    for (int nc = 0; nc < NC_; nc += 32) {
#pragma unroll
        for (int sub = 0; sub < 2; ++sub) {
            // k B-fragments for the 16-col n-subtile, K = DK split into 2x32
            Frag kf0, kf1;
            {
                int krow = nc + sub * 16 + m16;
                const ushort_t* kp = k + qkBase + (long)krow * DK_;
                kf0.u[0] = *(const uint4*)&kp[kh * 8];
                kf0.u[1] = *(const uint4*)&kp[16 + kh * 8];
                kf1.u[0] = *(const uint4*)&kp[32 + kh * 8];
                kf1.u[1] = *(const uint4*)&kp[48 + kh * 8];
            }
            v8f at = vz;
            at = wmma_bf16(qf[0], kf0, at);
            at = wmma_bf16(qf[1], kf1, at);

            // Cn = C*f + attn*i ; stream Cn out; stash bf16 for second GEMM
#pragma unroll
            for (int r = 0; r < 8; ++r) {
                int dg = d0 + wv * 16 + r + 8 * kh;
                int ng = nc + sub * 16 + m16;
                long o = ifBase + (long)dg * NC_ + ng;
                float cn = C[o] * bf2f(fact[o]) + at[r] * bf2f(iact[o]);
                Cn[o] = cn;
                cnS[wv][r + 8 * kh][sub * 16 + m16] = f2bf(cn);
            }
        }

        // out += Cn(16x32) @ v(32x64); A-frag from wave-private LDS, B from vT
        Frag cf;
        cf.u[0] = *(const uint4*)&cnS[wv][m16][kh * 8];
        cf.u[1] = *(const uint4*)&cnS[wv][m16][16 + kh * 8];
#pragma unroll
        for (int cb = 0; cb < 4; ++cb) {
            Frag wf;
            const ushort_t* vp = vT + (long)bh * DK_ * NC_
                               + (long)(cb * 16 + m16) * NC_ + nc;
            wf.u[0] = *(const uint4*)&vp[kh * 8];
            wf.u[1] = *(const uint4*)&vp[16 + kh * 8];
            outAcc[cb] = wmma_bf16(cf, wf, outAcc[cb]);
        }
    }

    // store out in (B, R, H*DK) bf16 for the o-projection GEMM
#pragma unroll
    for (int cb = 0; cb < 4; ++cb)
#pragma unroll
        for (int r = 0; r < 8; ++r) {
            int dg = d0 + wv * 16 + r + 8 * kh;
            int e  = cb * 16 + m16;
            long dst = (((long)b * NC_ + dg) * H_ + hh) * DK_ + e;
            outB[dst] = f2bf(outAcc[cb][r]);
        }
}

// ---------------------------------------------------------------------------
// Host side
// ---------------------------------------------------------------------------

struct ParamIdx { int w_ih, w_hh, bias, iv, ig, ib, fv, fg, fb,
                      qv, qg, qb, kv, kg, kb, vv, vg, vb, ov, og, ob; };

extern "C" void kernel_launch(void* const* d_in, const int* in_sizes, int n_in,
                              void* d_out, int out_size, void* d_ws, size_t ws_size,
                              hipStream_t stream) {
    (void)n_in; (void)out_size; (void)ws_size;

    // --- input index maps: insertion order vs jax-pytree (alphabetical) ---
    const bool alpha = (in_sizes[0] == B_ * H_ * 1024 * 1024); // C1 first?
    int iIn1, iIn2, iH1, iC1, iH2, iC2, pbase[2];
    ParamIdx po;
    if (!alpha) {
        iIn1 = 0; iIn2 = 1; iH1 = 2; iC1 = 3; iH2 = 4; iC2 = 5;
        pbase[0] = 6; pbase[1] = 27;
        po = { 0,1,2, 3,4,5, 6,7,8, 9,10,11, 12,13,14, 15,16,17, 18,19,20 };
    } else {
        iC1 = 0; iC2 = 1; iH1 = 2; iH2 = 3; iIn1 = 4; iIn2 = 5;
        pbase[0] = 6; pbase[1] = 27;
        po.bias = 0; po.fb = 1; po.fg = 2; po.fv = 3; po.ib = 4; po.ig = 5; po.iv = 6;
        po.kb = 7; po.kg = 8; po.kv = 9; po.ob = 10; po.og = 11; po.ov = 12;
        po.qb = 13; po.qg = 14; po.qv = 15; po.vb = 16; po.vg = 17; po.vv = 18;
        po.w_hh = 19; po.w_ih = 20;
    }

    // --- workspace carve-up (bytes, 256-aligned) ---
    size_t off = 0;
    auto alloc = [&](size_t bytes) -> char* {
        char* p = (char*)d_ws + off;
        off = (off + bytes + 255) & ~(size_t)255;
        return p;
    };
    const size_t szBig   = (size_t)8192 * 512 * 2;
    const size_t szGate  = (size_t)1536 * 512 * 2;
    const size_t szSmall = (size_t)512  * 512 * 2;
    const size_t szAct   = (size_t)M_ * 512 * 2;
    ushort_t* Wi_b  = (ushort_t*)alloc(szBig);
    ushort_t* Wf_b  = (ushort_t*)alloc(szBig);
    ushort_t* Wih_b = (ushort_t*)alloc(szGate);
    ushort_t* Whh_b = (ushort_t*)alloc(szGate);
    ushort_t* Wq_b  = (ushort_t*)alloc(szSmall);
    ushort_t* Wk_b  = (ushort_t*)alloc(szSmall);
    ushort_t* Wv_b  = (ushort_t*)alloc(szSmall);
    ushort_t* Wo_b  = (ushort_t*)alloc(szSmall);
    ushort_t* x1b   = (ushort_t*)alloc(szAct);
    ushort_t* x2b   = (ushort_t*)alloc(szAct);
    ushort_t* h1b   = (ushort_t*)alloc(szAct);
    ushort_t* h2b   = (ushort_t*)alloc(szAct);
    float*    gates = (float*)   alloc((size_t)M_ * 1536 * 4);
    ushort_t* gi_b  = (ushort_t*)alloc(szAct);
    ushort_t* gf_b  = (ushort_t*)alloc(szAct);
    float*    osig  = (float*)   alloc((size_t)M_ * 512 * 4);
    ushort_t* iact  = (ushort_t*)alloc((size_t)B_ * H_ * NC_ * NC_ * 2);
    ushort_t* fact  = (ushort_t*)alloc((size_t)B_ * H_ * NC_ * NC_ * 2);
    ushort_t* q_b   = (ushort_t*)alloc((size_t)B_ * H_ * NC_ * DK_ * 2);
    ushort_t* k_b   = (ushort_t*)alloc((size_t)B_ * H_ * NC_ * DK_ * 2);
    ushort_t* v_b   = (ushort_t*)alloc((size_t)B_ * H_ * NC_ * DK_ * 2);
    ushort_t* out_b = (ushort_t*)alloc(szAct);

    // --- one-time activation conversions ---
    const int nAct = M_ * 512;
    f2bf_kernel<<<(nAct + 255) / 256, 256, 0, stream>>>((const float*)d_in[iIn1], x1b, nAct);
    f2bf_kernel<<<(nAct + 255) / 256, 256, 0, stream>>>((const float*)d_in[iIn2], x2b, nAct);
    f2bf_kernel<<<(nAct + 255) / 256, 256, 0, stream>>>((const float*)d_in[iH1],  h1b, nAct);
    f2bf_kernel<<<(nAct + 255) / 256, 256, 0, stream>>>((const float*)d_in[iH2],  h2b, nAct);

    const long hoElems = (long)M_ * 512;
    const long cnElems = (long)B_ * H_ * NC_ * NC_;
    float* out = (float*)d_out;

    for (int c = 0; c < 2; ++c) {
        const int pb = pbase[c];
        auto P = [&](int o) -> const float* { return (const float*)d_in[pb + o]; };
        const ushort_t* xa  = c ? x2b : x1b;   // cell's x1
        const ushort_t* xb2 = c ? x1b : x2b;   // cell's x2
        const ushort_t* hb  = c ? h2b : h1b;
        const float* Cin = (const float*)d_in[c ? iC2 : iC1];
        float* hoOut = out + (c ? (hoElems + cnElems) : 0);
        float* CnOut = out + (c ? (2 * hoElems + cnElems) : hoElems);

        // --- weight prep (weight-norm folded into bf16 weights) ---
        wn_prep_kernel<<<8192, 256, 0, stream>>>(P(po.iv), P(po.ig), Wi_b, 512);
        wn_prep_kernel<<<8192, 256, 0, stream>>>(P(po.fv), P(po.fg), Wf_b, 512);
        wn_prep_kernel<<< 512, 256, 0, stream>>>(P(po.qv), P(po.qg), Wq_b, 512);
        wn_prep_kernel<<< 512, 256, 0, stream>>>(P(po.kv), P(po.kg), Wk_b, 512);
        wn_prep_kernel<<< 512, 256, 0, stream>>>(P(po.vv), P(po.vg), Wv_b, 512);
        wn_prep_kernel<<< 512, 256, 0, stream>>>(P(po.ov), P(po.og), Wo_b, 512);
        wn_prep_kernel<<<1536, 256, 0, stream>>>(P(po.w_ih), nullptr, Wih_b, 512);
        wn_prep_kernel<<<1536, 256, 0, stream>>>(P(po.w_hh), nullptr, Whh_b, 512);

        // --- gates = x1 @ Wih^T + bias + h @ Whh^T  (fp32, M_ x 1536) ---
        dim3 gGate(1536 / 128, M_ / 128);
        gemm_wmma_kernel<<<gGate, 256, 0, stream>>>(
            xa, 512, Wih_b, 512, M_, 1536, 512,
            P(po.bias), nullptr, 1.f, 0, 0, gates, nullptr, 1536, 1, 1);
        gemm_wmma_kernel<<<gGate, 256, 0, stream>>>(
            hb, 512, Whh_b, 512, M_, 1536, 512,
            nullptr, nullptr, 1.f, 0, 1, gates, nullptr, 1536, 1, 1);

        split_gates_kernel<<<(nAct + 255) / 256, 256, 0, stream>>>(gates, gi_b, gf_b, osig, nAct);

        // --- i = tanh(wn(gate_i)) , f = sigmoid(wn(gate_f)) -> (B,H,R,NC) bf16 ---
        dim3 gBig(8192 / 128, M_ / 128);
        gemm_wmma_kernel<<<gBig, 256, 0, stream>>>(
            gi_b, 512, Wi_b, 512, M_, 8192, 512,
            P(po.ib), nullptr, 1.f, 2, 0, nullptr, iact, 0, NC_, NC_);
        gemm_wmma_kernel<<<gBig, 256, 0, stream>>>(
            gf_b, 512, Wf_b, 512, M_, 8192, 512,
            P(po.fb), nullptr, 1.f, 3, 0, nullptr, fact, 0, NC_, NC_);

        // --- q (pre-scaled), k -> (B,H,R,DK);  v -> (B,H,DK,R) transposed ---
        dim3 gQkv(512 / 128, M_ / 128);
        gemm_wmma_kernel<<<gQkv, 256, 0, stream>>>(
            xa, 512, Wq_b, 512, M_, 512, 512,
            P(po.qb), nullptr, SCALE_, 4, 0, nullptr, q_b, 0, DK_, NC_);
        gemm_wmma_kernel<<<gQkv, 256, 0, stream>>>(
            xb2, 512, Wk_b, 512, M_, 512, 512,
            P(po.kb), nullptr, 1.f, 4, 0, nullptr, k_b, 0, DK_, NC_);
        gemm_wmma_kernel<<<gQkv, 256, 0, stream>>>(
            xb2, 512, Wv_b, 512, M_, 512, 512,
            P(po.vb), nullptr, 1.f, 5, 0, nullptr, v_b, 0, DK_, NC_);

        // --- fused: attn, Cn update (-> d_out), Cn @ v ---
        attn_fused_kernel<<<dim3(NC_ / 64, B_ * H_), 128, 0, stream>>>(
            q_b, k_b, v_b, iact, fact, Cin, CnOut, out_b);

        // --- ho = sigmoid(gate_o) * (out @ Wo^T + ob) -> d_out fp32 ---
        gemm_wmma_kernel<<<gQkv, 256, 0, stream>>>(
            out_b, 512, Wo_b, 512, M_, 512, 512,
            P(po.ob), osig, 1.f, 0, 0, hoOut, nullptr, 512, 1, 1);
    }
}
